// LlamaFlexAttention_24696061952631
// MI455X (gfx1250) — matrix operand order repaired and checked by reading, not compiled, not run on hardware
//
#include <hip/hip_runtime.h>
#include <hip/hip_bf16.h>

typedef __attribute__((ext_vector_type(16))) __bf16 v16bf;
typedef __attribute__((ext_vector_type(8)))  float  v8f;

union FragA { v16bf v; unsigned u[8]; uint4 q[2]; };
union FragF { v8f v; float f[8]; };

__device__ __forceinline__ unsigned short f2bf(float f) {
    union { float f; unsigned u; } c; c.f = f;
    unsigned u = c.u;
    unsigned r = u + 0x7FFFu + ((u >> 16) & 1u);   // round-to-nearest-even
    return (unsigned short)(r >> 16);
}

// wave-level fence: stop compiler reordering LDS ops (same-wave LDS is in-order)
__device__ __forceinline__ void wave_fence() {
    __builtin_amdgcn_wave_barrier();
    asm volatile("" ::: "memory");
}

// ---------------------------------------------------------------------------
// seq_len = sum(attention_mask) - LCK
// ---------------------------------------------------------------------------
__global__ void seqlen_kernel(const int* __restrict__ mask, int* __restrict__ out) {
    __shared__ int sdata[256];
    int t = threadIdx.x, s = 0;
    for (int i = t; i < 1024; i += 256) s += mask[i];
    sdata[t] = s; __syncthreads();
    for (int off = 128; off > 0; off >>= 1) {
        if (t < off) sdata[t] += sdata[t + off];
        __syncthreads();
    }
    if (t == 0) out[0] = sdata[0] - 2;
}

// ---------------------------------------------------------------------------
// f32 -> bf16 convert with column placement into a (possibly fused) dst.
// ---------------------------------------------------------------------------
__global__ void convert_bf16_kernel(const float* __restrict__ src,
                                    unsigned short* __restrict__ dst,
                                    int logc, int dstld, int dstoff) {
    int idx = blockIdx.x * 256 + threadIdx.x;
    int r = idx >> logc, c = idx & ((1 << logc) - 1);
    dst[(size_t)r * dstld + dstoff + c] = f2bf(src[idx]);
}

// ---------------------------------------------------------------------------
// GEMM: C[M x N](f32) = A[M x K](bf16) @ B[K x N](bf16).
// 128x128 block tile, BK=32, 8 waves. Double-buffered LDS:
//   A tile: GLOBAL_LOAD_ASYNC_TO_LDS_B128 (ASYNCcnt), 80B row stride
//   B tile: uint4 global loads, transposed to [n][k], 80B row stride
// One barrier per K-step; next tile's loads overlap current tile's 8 wmmas.
// ---------------------------------------------------------------------------
static __device__ __forceinline__ void issue_tileA(const unsigned short* A, int m0,
                                                   int K, int k0, int tid,
                                                   unsigned lbase) {
    for (int i = 0; i < 2; ++i) {
        int cid = tid + 256 * i;
        int r = cid >> 2, c16 = cid & 3;               // row, 16B chunk in row
        unsigned voff = (unsigned)(((size_t)(m0 + r) * K + k0 + c16 * 8) * 2);
        unsigned loff = lbase + (unsigned)(r * 80 + c16 * 16);
        asm volatile("global_load_async_to_lds_b128 %0, %1, %2"
                     :: "v"(loff), "v"(voff), "s"(A) : "memory");
    }
}
static __device__ __forceinline__ void load_tileB(const unsigned short* B, int n0,
                                                  int ldb, int k0, int tid,
                                                  unsigned short* bs) {
    for (int i = 0; i < 2; ++i) {
        int cid = tid + 256 * i;
        int kk = cid >> 4, c8 = cid & 15;              // row, 8-elem chunk
        const uint4 g = *(const uint4*)(B + (size_t)(k0 + kk) * ldb + n0 + c8 * 8);
        unsigned gu[4] = {g.x, g.y, g.z, g.w};
        for (int j = 0; j < 4; ++j) {
            bs[(c8 * 8 + 2 * j) * 40 + kk]     = (unsigned short)(gu[j] & 0xFFFFu);
            bs[(c8 * 8 + 2 * j + 1) * 40 + kk] = (unsigned short)(gu[j] >> 16);
        }
    }
}

__global__ __launch_bounds__(256) void gemm_bb_kernel(
    const unsigned short* __restrict__ A, const unsigned short* __restrict__ B,
    float* __restrict__ C, int K, int ldb, int ldc) {
    __shared__ unsigned       AsU[2][128 * 20];   // 80B row stride, 16B aligned
    __shared__ unsigned short Bs[2][128 * 40];    // [n][k], 80B row stride

    const int m0 = blockIdx.y * 128, n0 = blockIdx.x * 128;
    const int tid = threadIdx.x, w = tid >> 5, lane = tid & 31;
    const int half = lane >> 4, l16 = lane & 15;
    const unsigned lbase0 = (unsigned)(size_t)(&AsU[0][0]);
    const unsigned lbase1 = (unsigned)(size_t)(&AsU[1][0]);

    FragF acc[8];
    for (int nt = 0; nt < 8; ++nt)
        for (int i = 0; i < 8; ++i) acc[nt].f[i] = 0.0f;

    const int niter = K >> 5;
    issue_tileA(A, m0, K, 0, tid, lbase0);
    load_tileB(B, n0, ldb, 0, tid, &Bs[0][0]);

    for (int it = 0; it < niter; ++it) {
        const int p = it & 1;
        asm volatile("s_wait_asynccnt 0x0" ::: "memory");
        __syncthreads();                        // tile `it` fully in LDS
        if (it + 1 < niter) {                   // prefetch tile it+1 under compute
            issue_tileA(A, m0, K, (it + 1) << 5, tid, p ? lbase0 : lbase1);
            load_tileB(B, n0, ldb, (it + 1) << 5, tid, &Bs[p ^ 1][0]);
        }
        FragA a;                                // A frag: 2 x ds_load_b128
        {
            const uint4* a4 = (const uint4*)(&AsU[p][(w * 16 + l16) * 20 + 4 * half]);
            a.q[0] = a4[0]; a.q[1] = a4[2];
        }
        for (int nt = 0; nt < 8; ++nt) {        // B frag: 2 x ds_load_b128 each
            FragA b;
            const uint4* b4 = (const uint4*)(&Bs[p][(nt * 16 + l16) * 40 + 8 * half]);
            b.q[0] = b4[0]; b.q[1] = b4[2];
            acc[nt].v = __builtin_amdgcn_wmma_f32_16x16x32_bf16(
                false, a.v, false, b.v, (short)0, acc[nt].v, false, false);
        }
    }
    for (int nt = 0; nt < 8; ++nt)
        for (int v = 0; v < 8; ++v)
            C[(size_t)(m0 + w * 16 + v + 8 * half) * ldc + n0 + nt * 16 + l16] = acc[nt].f[v];
}

// ---------------------------------------------------------------------------
// Past KV: fp32 -> bf16; keys row-major [KVH][3072][64], values transposed.
// ---------------------------------------------------------------------------
__global__ void past_kv_kernel(const float* __restrict__ pk, const float* __restrict__ pv,
                               unsigned short* __restrict__ keyc,
                               unsigned short* __restrict__ vT) {
    int idx = blockIdx.x * 256 + threadIdx.x;       // 1048576
    int hk = idx >> 17, r = idx & 131071, p = r >> 6, d = r & 63;
    keyc[((size_t)hk * 3072 + p) * 64 + d] = f2bf(pk[idx]);
    vT[((size_t)hk * 64 + d) * 3072 + p]   = f2bf(pv[idx]);
}

// ---------------------------------------------------------------------------
// RoPE on Q -> bf16 [H][Q][D]
// ---------------------------------------------------------------------------
__global__ void rope_q_kernel(const float* __restrict__ qkv, const int* __restrict__ pid,
                              unsigned short* __restrict__ qbf) {
    int idx = blockIdx.x * 256 + threadIdx.x;       // 1024*2048
    int q = idx >> 11, c = idx & 2047, h = c >> 6, d = c & 63;
    float x  = qkv[(size_t)q * 3072 + c];
    int  dp  = (d < 32) ? d + 32 : d - 32;
    float xp = qkv[(size_t)q * 3072 + h * 64 + dp];
    float rot = (d < 32) ? -xp : xp;
    float pos = (float)(pid[q] + 2);
    float inv = __expf(-(float)(d & 31) * 0.28782313662424256f); // ln(1e4)/32
    float ang = pos * inv;
    qbf[((size_t)h * 1024 + q) * 64 + d] = f2bf(x * __cosf(ang) + rot * __sinf(ang));
}

// ---------------------------------------------------------------------------
// RoPE on new K (cache offset 2048) + scatter new V transposed
// ---------------------------------------------------------------------------
__global__ void rope_kv_kernel(const float* __restrict__ qkv, const int* __restrict__ pid,
                               unsigned short* __restrict__ keyc,
                               unsigned short* __restrict__ vT) {
    int idx = blockIdx.x * 256 + threadIdx.x;       // 1024*512
    int q = idx >> 9, c = idx & 511, hk = c >> 6, d = c & 63;
    size_t base = (size_t)q * 3072;
    float kx  = qkv[base + 2048 + c];
    int  dp   = (d < 32) ? d + 32 : d - 32;
    float kxp = qkv[base + 2048 + hk * 64 + dp];
    float rot = (d < 32) ? -kxp : kxp;
    float pos = (float)(pid[q] + 2);
    float inv = __expf(-(float)(d & 31) * 0.28782313662424256f);
    float ang = pos * inv;
    keyc[((size_t)hk * 3072 + 2048 + q) * 64 + d] =
        f2bf(kx * __cosf(ang) + rot * __sinf(ang));
    vT[((size_t)hk * 64 + d) * 3072 + 2048 + q] = f2bf(qkv[base + 2560 + c]);
}

// ---------------------------------------------------------------------------
// Flash attention, eagle3 mask. Block = (head, 128 q rows), 8 waves x 16 rows.
// ---------------------------------------------------------------------------
__global__ __launch_bounds__(256) void attn_kernel(
    const unsigned* __restrict__ qbf, const unsigned* __restrict__ keyc,
    const unsigned* __restrict__ vT, const int* __restrict__ seqlen_p,
    unsigned short* __restrict__ attnout) {
    constexpr int Q = 1024, D = 64, KV = 3072, LCK = 2;
    __shared__ unsigned short Plds[8 * 16 * 40];    // per-wave 16x32 probs, 80B rows

    const int qblk = blockIdx.x, h = blockIdx.y, kvh = h >> 2;
    const int tid = threadIdx.x, w = tid >> 5, lane = tid & 31;
    const int half = lane >> 4, l16 = lane & 15;
    const int q0w = qblk * 128 + w * 16;
    const int seq_len = seqlen_p[0];
    unsigned short* Pw = Plds + w * 16 * 40;

    // Q fragments (held entire sweep): d 0..31 / d 32..63 -- b128 loads
    FragA aq[2];
    {
        const unsigned* qrow = qbf + ((size_t)h * Q + q0w + l16) * 32;
        const uint4* q0 = (const uint4*)(qrow + 4 * half);
        const uint4* q1 = (const uint4*)(qrow + 16 + 4 * half);
        aq[0].q[0] = q0[0]; aq[0].q[1] = q0[2];
        aq[1].q[0] = q1[0]; aq[1].q[1] = q1[2];
    }

    float mrow[8], lrow[8];
    FragF o[4];
    for (int v = 0; v < 8; ++v) { mrow[v] = -1e30f; lrow[v] = 0.0f; }
    for (int dt = 0; dt < 4; ++dt)
        for (int v = 0; v < 8; ++v) o[dt].f[v] = 0.0f;

    for (int s = 0; s < 3; ++s) {
        int jc = qblk * 128 + 126 + s;              // qlast + 1 - (LCK - s)
        int jcount = (seq_len < jc) ? seq_len : jc;
        if (jcount <= 0) continue;
        int steps = (jcount + 31) >> 5;
        for (int t = 0; t < steps; ++t) {
            int kvbase = s * Q + t * 32;
            FragF sc[2];
            for (int nt = 0; nt < 2; ++nt)
                for (int v = 0; v < 8; ++v) sc[nt].f[v] = 0.0f;

            for (int nt = 0; nt < 2; ++nt) {
                const unsigned* krow =
                    keyc + ((size_t)kvh * KV + kvbase + nt * 16 + l16) * 32;
                FragA b0, b1;
                const uint4* k0 = (const uint4*)(krow + 4 * half);
                const uint4* k1 = (const uint4*)(krow + 16 + 4 * half);
                b0.q[0] = k0[0]; b0.q[1] = k0[2];
                b1.q[0] = k1[0]; b1.q[1] = k1[2];
                sc[nt].v = __builtin_amdgcn_wmma_f32_16x16x32_bf16(
                    false, aq[0].v, false, b0.v, (short)0, sc[nt].v, false, false);
                sc[nt].v = __builtin_amdgcn_wmma_f32_16x16x32_bf16(
                    false, aq[1].v, false, b1.v, (short)0, sc[nt].v, false, false);
            }
            // scale + eagle3 mask
            for (int nt = 0; nt < 2; ++nt)
                for (int v = 0; v < 8; ++v) {
                    int qg = q0w + v + 8 * half;
                    int j  = t * 32 + nt * 16 + l16;
                    bool ok = (j < seq_len) && (j <= qg - (LCK - s));
                    float x = sc[nt].f[v] * 0.125f;
                    sc[nt].f[v] = ok ? x : -1e30f;
                }
            // online softmax (N spans 16-lane halves)
            float alpha[8];
            for (int v = 0; v < 8; ++v) {
                float x = fmaxf(sc[0].f[v], sc[1].f[v]);
                for (int off = 1; off <= 8; off <<= 1)
                    x = fmaxf(x, __shfl_xor(x, off, 32));
                float nm = fmaxf(mrow[v], x);
                alpha[v] = __expf(mrow[v] - nm);
                mrow[v] = nm;
            }
            for (int nt = 0; nt < 2; ++nt)
                for (int v = 0; v < 8; ++v)
                    sc[nt].f[v] = __expf(sc[nt].f[v] - mrow[v]);
            for (int v = 0; v < 8; ++v) {
                float sum = sc[0].f[v] + sc[1].f[v];
                for (int off = 1; off <= 8; off <<= 1)
                    sum += __shfl_xor(sum, off, 32);
                lrow[v] = lrow[v] * alpha[v] + sum;
            }
            for (int dt = 0; dt < 4; ++dt)
                for (int v = 0; v < 8; ++v) o[dt].f[v] *= alpha[v];

            // C-layout probs -> wave-private LDS (bf16) -> A-layout (b128 reads)
            wave_fence();
            for (int nt = 0; nt < 2; ++nt)
                for (int v = 0; v < 8; ++v)
                    Pw[(v + 8 * half) * 40 + l16 + 16 * nt] = f2bf(sc[nt].f[v]);
            wave_fence();
            FragA ap;
            {
                const uint4* p4 = (const uint4*)(&Pw[l16 * 40 + 8 * half]);
                ap.q[0] = p4[0]; ap.q[1] = p4[2];
            }
            // P @ V: V^T rows contiguous over kv -> b128 loads
            for (int dt = 0; dt < 4; ++dt) {
                FragA bv;
                const unsigned* vrow =
                    vT + (((size_t)(kvh * D + dt * 16 + l16)) * KV + kvbase) / 2;
                const uint4* v4 = (const uint4*)(vrow + 4 * half);
                bv.q[0] = v4[0]; bv.q[1] = v4[2];
                o[dt].v = __builtin_amdgcn_wmma_f32_16x16x32_bf16(
                    false, ap.v, false, bv.v, (short)0, o[dt].v, false, false);
            }
        }
    }
    for (int dt = 0; dt < 4; ++dt)
        for (int v = 0; v < 8; ++v) {
            int q = q0w + v + 8 * half;
            attnout[(size_t)q * 2048 + h * 64 + dt * 16 + l16] =
                f2bf(o[dt].f[v] / lrow[v]);
        }
}

// ---------------------------------------------------------------------------
extern "C" void kernel_launch(void* const* d_in, const int* in_sizes, int n_in,
                              void* d_out, int out_size, void* d_ws, size_t ws_size,
                              hipStream_t stream) {
    (void)in_sizes; (void)n_in; (void)out_size; (void)ws_size;
    const float* hs  = (const float*)d_in[0];
    const float* pk  = (const float*)d_in[1];
    const float* pv  = (const float*)d_in[2];
    const int*   am  = (const int*)  d_in[3];
    const int*   pid = (const int*)  d_in[4];
    const float* Wq  = (const float*)d_in[5];
    const float* Wk  = (const float*)d_in[6];
    const float* Wv  = (const float*)d_in[7];
    const float* Wo  = (const float*)d_in[8];

    char* ws = (char*)d_ws;
    size_t off = 0;
    float*          qkv     = (float*)          (ws + off); off += 12582912; // Q x 3072 f32
    unsigned short* qbf     = (unsigned short*) (ws + off); off += 4194304;  // H,Q,D bf16
    unsigned short* keyc    = (unsigned short*) (ws + off); off += 3145728;  // KVH,3072,D
    unsigned short* vT      = (unsigned short*) (ws + off); off += 3145728;  // KVH,D,3072
    unsigned short* attnbf  = (unsigned short*) (ws + off); off += 4194304;  // Q x 2048 bf16
    unsigned short* hsbf    = (unsigned short*) (ws + off); off += 8388608;  // Q x 4096 bf16
    unsigned short* wqkvbf  = (unsigned short*) (ws + off); off += 25165824; // 4096 x 3072
    unsigned short* wobf    = (unsigned short*) (ws + off); off += 8388608;  // 2048 x 2048
    int*            sl      = (int*)            (ws + off);

    seqlen_kernel<<<1, 256, 0, stream>>>(am, sl);

    // one-time bf16 conversions (weights fused into [4096 x 3072])
    convert_bf16_kernel<<<4194304 / 256, 256, 0, stream>>>(hs, hsbf, 12, 4096, 0);
    convert_bf16_kernel<<<8388608 / 256, 256, 0, stream>>>(Wq, wqkvbf, 11, 3072, 0);
    convert_bf16_kernel<<<2097152 / 256, 256, 0, stream>>>(Wk, wqkvbf, 9, 3072, 2048);
    convert_bf16_kernel<<<2097152 / 256, 256, 0, stream>>>(Wv, wqkvbf, 9, 3072, 2560);
    convert_bf16_kernel<<<4194304 / 256, 256, 0, stream>>>(Wo, wobf, 11, 2048, 0);

    // fused QKV projection: [1024 x 4096] @ [4096 x 3072] -> qkv (f32)
    gemm_bb_kernel<<<dim3(24, 8), 256, 0, stream>>>(hsbf, wqkvbf, qkv, 4096, 3072, 3072);

    past_kv_kernel<<<1048576 / 256, 256, 0, stream>>>(pk, pv, keyc, vT);
    rope_q_kernel <<<2097152 / 256, 256, 0, stream>>>(qkv, pid, qbf);
    rope_kv_kernel<<< 524288 / 256, 256, 0, stream>>>(qkv, pid, keyc, vT);

    attn_kernel<<<dim3(8, 32), 256, 0, stream>>>(
        (const unsigned*)qbf, (const unsigned*)keyc, (const unsigned*)vT, sl, attnbf);

    // output projection: [1024 x 2048] @ [2048 x 2048] -> d_out (f32)
    gemm_bb_kernel<<<dim3(16, 8), 256, 0, stream>>>(attnbf, wobf, (float*)d_out,
                                                    2048, 2048, 2048);
}
